// SLSTM_SW_43284680409163
// MI455X (gfx1250) — matrix operand order repaired and checked by reading, not compiled, not run on hardware
//
#include <hip/hip_runtime.h>
#include <hip/hip_bf16.h>

typedef __attribute__((ext_vector_type(16))) __bf16 v16bf;
typedef __attribute__((ext_vector_type(8)))  __bf16 v8bf;
typedef __attribute__((ext_vector_type(8)))  float  v8f;

#if defined(__gfx1250__)
#if __has_builtin(__builtin_amdgcn_global_load_async_to_lds_b128)
#define USE_ASYNC_COPY 1
#endif
#endif
#ifndef USE_ASYNC_COPY
#define USE_ASYNC_COPY 0
#endif

namespace {
constexpr int BATCH = 32;
constexpr int SEQ   = 512;
constexpr int WD    = 512;
constexpr int HD    = 512;
constexpr long NROW = (long)BATCH * SEQ;   // 16384
constexpr int KA    = 2560;                // 512(h) + 1024(hb|ha) + 512(emb) + 512(sw_h)
constexpr int NC    = 4608;                // 9 * 512 (8 gates + gf column block)
constexpr int KSTEPS = KA / 32;            // 80
constexpr int ASTR  = 40;                  // LDS row stride (bf16 elems), 80B, 16B aligned
constexpr int BSTR  = 40;
}

__device__ __forceinline__ float sigmoidf_(float x) { return 1.0f / (1.0f + __expf(-x)); }

__device__ __forceinline__ void wait_async0() {
#if __has_builtin(__builtin_amdgcn_s_wait_asynccnt)
  __builtin_amdgcn_s_wait_asynccnt(0);
#else
  asm volatile("s_wait_asynccnt 0x0" ::: "memory");
#endif
}

// ---------------------------------------------------------------------------
// Pack weights into fragment-native bf16 layout: Bpack[(k>>5)*NC + col][k&31]
// col = g*512 + j ; g<8 -> {Wx,Wh,Wi,Ws}[g], g==8 -> gWhf (=gW[5]) in K<512
// ---------------------------------------------------------------------------
__global__ __launch_bounds__(256) void pack_weights(
    const float* __restrict__ Wx, const float* __restrict__ Wh,
    const float* __restrict__ Wi, const float* __restrict__ Ws,
    const float* __restrict__ gW, __bf16* __restrict__ Bpack)
{
  long idx = (long)blockIdx.x * 256 + threadIdx.x;     // over KA*NC
  if (idx >= (long)KA * NC) return;
  int k  = (int)(idx / NC);
  int co = (int)(idx % NC);
  int g = co >> 9, j = co & 511;
  float v = 0.0f;
  if (g < 8) {
    if (k < 512)       v = Wx[((long)g * 512  +  k        ) * 512 + j];
    else if (k < 1536) v = Wh[((long)g * 1024 + (k - 512 )) * 512 + j];
    else if (k < 2048) v = Wi[((long)g * 512  + (k - 1536)) * 512 + j];
    else               v = Ws[((long)g * 512  + (k - 2048)) * 512 + j];
  } else if (k < 512) {
    v = gW[(5L * 512 + k) * 512 + j];                  // gWhf
  }
  Bpack[(((long)(k >> 5) * NC + co) << 5) + (k & 31)] = (__bf16)v;
}

// h/c init with sequence mask
__global__ __launch_bounds__(256) void init_hc(
    const float* __restrict__ ih, const float* __restrict__ ic,
    const int* __restrict__ mask, float* __restrict__ h, float* __restrict__ c)
{
  long idx = (long)blockIdx.x * 256 + threadIdx.x;     // NROW*HD
  long n = idx >> 9;
  float m = (float)mask[n];
  h[idx] = ih[idx] * m;
  c[idx] = ic[idx] * m;
}

// emb block of A (constant across layers): cols [1536,2048)
__global__ __launch_bounds__(256) void init_emb(
    const float* __restrict__ w, const int* __restrict__ mask, __bf16* __restrict__ A)
{
  long idx = (long)blockIdx.x * 256 + threadIdx.x;     // NROW*WD
  long n = idx >> 9; int j = (int)(idx & 511);
  float m = (float)mask[n];
  A[n * KA + 1536 + j] = (__bf16)(w[idx] * m);
}

// out[b,j] = mean over S of x[b,s,j]
__global__ __launch_bounds__(256) void mean_bh(const float* __restrict__ x, float* __restrict__ out)
{
  int idx = blockIdx.x * 256 + threadIdx.x;            // BATCH*HD
  int b = idx >> 9, j = idx & 511;
  const float* p = x + ((long)b * SEQ) * HD + j;
  float s = 0.0f;
  for (int t = 0; t < SEQ; ++t) s += p[(long)t * HD];
  out[idx] = s * (1.0f / SEQ);
}

// Build bf16 A blocks from current h: [h | hb | ha | (emb) | sw_h]
__global__ __launch_bounds__(256) void prep_A(
    const float* __restrict__ h, const int* __restrict__ pos, __bf16* __restrict__ A)
{
  long idx = (long)blockIdx.x * 256 + threadIdx.x;     // NROW*HD
  long n = idx >> 9; int j = (int)(idx & 511);
  int b = (int)(n >> 9), s = (int)(n & 511);
  float hv = h[idx];
  float hb = 0.0f, ha = 0.0f;
  if (s >= 1)        hb += h[idx - HD];
  if (s >= 2)        hb += h[idx - 2 * HD];
  if (s <= SEQ - 2)  ha += h[idx + HD];
  if (s <= SEQ - 3)  ha += h[idx + 2 * HD];
  int p = pos[n];
  float sw = (p == 0) ? 0.0f : h[((long)b * SEQ + (p - 1)) * HD + j];
  __bf16* ar = A + n * KA;
  ar[j]        = (__bf16)hv;
  ar[512 + j]  = (__bf16)hb;
  ar[1024 + j] = (__bf16)ha;
  ar[2048 + j] = (__bf16)sw;
}

// Per-batch small matmuls: gd, go, biasExt[b][g][j]
// biasExt g<8: b[g,j] + dummy_h@Wd[g];  g==8: gbf[j] + dummy_h@gWxf (=gW[4])
__global__ __launch_bounds__(512) void small_gates(
    const float* __restrict__ dh, const float* __restrict__ comb,
    const float* __restrict__ gW, const float* __restrict__ gb,
    const float* __restrict__ bvec, const float* __restrict__ Wd,
    float* __restrict__ gd, float* __restrict__ go, float* __restrict__ biasE)
{
  __shared__ float sdh[512];
  __shared__ float scb[512];
  int b = blockIdx.x, j = threadIdx.x;
  sdh[j] = dh[b * 512 + j];
  scb[j] = comb[b * 512 + j];
  __syncthreads();
  float a_gd = gb[j], a_go = gb[512 + j], a_bf = gb[1024 + j];
  float a_bias[8];
#pragma unroll
  for (int g = 0; g < 8; ++g) a_bias[g] = bvec[g * 512 + j];
  for (int k = 0; k < 512; ++k) {
    float dv = sdh[k], cv = scb[k];
    a_gd += dv * gW[(0L * 512 + k) * 512 + j] + cv * gW[(1L * 512 + k) * 512 + j];
    a_go += dv * gW[(2L * 512 + k) * 512 + j] + cv * gW[(3L * 512 + k) * 512 + j];
    a_bf += dv * gW[(4L * 512 + k) * 512 + j];
#pragma unroll
    for (int g = 0; g < 8; ++g) a_bias[g] += dv * Wd[((long)g * 512 + k) * 512 + j];
  }
  gd[b * 512 + j] = sigmoidf_(a_gd);
  go[b * 512 + j] = sigmoidf_(a_go);
  float* be = biasE + (long)b * 9 * 512;
#pragma unroll
  for (int g = 0; g < 8; ++g) be[g * 512 + j] = a_bias[g];
  be[8 * 512 + j] = a_bf;
}

// ---------------------------------------------------------------------------
// Main GEMM: Z[N x 4608] = A[N x 2560] (bf16) * Bpack (bf16) + biasE
// 128x128 tile, 8 waves (4M x 2N), wave tile 32x64, K-step 32, bf16 WMMA.
// Global->LDS staging via GLOBAL_LOAD_ASYNC_TO_LDS_B128 (ASYNCcnt) when the
// toolchain declares the builtin; register staging fallback otherwise.
// ---------------------------------------------------------------------------
__global__ __launch_bounds__(256) void gemm_wmma(
    const __bf16* __restrict__ A, const __bf16* __restrict__ Bp,
    const float* __restrict__ biasE, float* __restrict__ Z)
{
  __shared__ __bf16 As[2][128 * ASTR];
  __shared__ __bf16 Bs[2][128 * BSTR];

  const int tid  = threadIdx.x;
  const int wave = tid >> 5;
  const int lane = tid & 31;
  const int waveM = wave & 3;      // 4 waves along M, 32 rows each
  const int waveN = wave >> 2;     // 2 waves along N, 64 cols each
  const int hl = lane & 15;
  const int hi = lane >> 4;        // 0 | 1

  const long rowBase = (long)blockIdx.x * 128;
  const int  colBase = blockIdx.y * 128;

  // global->LDS staging: each thread moves 32B of A and 32B of B per k-step
  const int ar = tid >> 1, ah = tid & 1;
  const __bf16* aG = A + (rowBase + ar) * (long)KA + ah * 16;
  const int bc = tid >> 1, bh = tid & 1;
  const __bf16* bG = Bp + (((long)colBase + bc) << 5) + bh * 16;
  const long bStep = (long)NC << 5;

  v8f acc[2][4] = {};

#if USE_ASYNC_COPY
  typedef int v4i_ __attribute__((vector_size(16)));
  typedef __attribute__((address_space(1))) v4i_* gq_t;   // 16B chunk in global
  typedef __attribute__((address_space(3))) v4i_* lq_t;   // 16B chunk in LDS

  __bf16* aD0 = &As[0][ar * ASTR + ah * 16];
  __bf16* aD1 = &As[1][ar * ASTR + ah * 16];
  __bf16* bD0 = &Bs[0][bc * BSTR + bh * 16];
  __bf16* bD1 = &Bs[1][bc * BSTR + bh * 16];

  auto issue = [&](int kb, int buf) {
    const __bf16* ap = aG + (long)kb * 32;
    const __bf16* bp = bG + (long)kb * bStep;
    __bf16* ad = buf ? aD1 : aD0;
    __bf16* bd = buf ? bD1 : bD0;
    __builtin_amdgcn_global_load_async_to_lds_b128((gq_t)ap,       (lq_t)ad,       0, 0);
    __builtin_amdgcn_global_load_async_to_lds_b128((gq_t)(ap + 8), (lq_t)(ad + 8), 0, 0);
    __builtin_amdgcn_global_load_async_to_lds_b128((gq_t)bp,       (lq_t)bd,       0, 0);
    __builtin_amdgcn_global_load_async_to_lds_b128((gq_t)(bp + 8), (lq_t)(bd + 8), 0, 0);
  };

  issue(0, 0);
  wait_async0();
  __syncthreads();
#else
  v8bf a0, a1, b0, b1;
  auto fetch = [&](int kb) {
    const __bf16* ap = aG + (long)kb * 32;
    a0 = *(const v8bf*)ap;
    a1 = *(const v8bf*)(ap + 8);
    const __bf16* bp = bG + (long)kb * bStep;
    b0 = *(const v8bf*)bp;
    b1 = *(const v8bf*)(bp + 8);
  };
  auto stage = [&](int buf) {
    __bf16* as = &As[buf][ar * ASTR + ah * 16];
    *(v8bf*)as = a0; *(v8bf*)(as + 8) = a1;
    __bf16* bs = &Bs[buf][bc * BSTR + bh * 16];
    *(v8bf*)bs = b0; *(v8bf*)(bs + 8) = b1;
  };
  fetch(0);
  stage(0);
  __syncthreads();
#endif

  for (int kb = 0; kb < KSTEPS; ++kb) {
    const int cur = kb & 1;
#if USE_ASYNC_COPY
    if (kb + 1 < KSTEPS) issue(kb + 1, cur ^ 1);
#else
    if (kb + 1 < KSTEPS) fetch(kb + 1);
    if (kb + 2 < KSTEPS) {
      __builtin_prefetch(aG + (long)(kb + 2) * 32, 0, 3);
      __builtin_prefetch(bG + (long)(kb + 2) * bStep, 0, 3);
    }
#endif

    // A fragments (16x32): lane<16 -> K{0..7,16..23}, lane>=16 -> K{8..15,24..31}
    v16bf af[2];
#pragma unroll
    for (int mi = 0; mi < 2; ++mi) {
      const __bf16* p = &As[cur][(waveM * 32 + mi * 16 + hl) * ASTR + hi * 8];
      v8bf lo  = *(const v8bf*)p;
      v8bf hi8 = *(const v8bf*)(p + 16);
      af[mi] = __builtin_shufflevector(lo, hi8, 0,1,2,3,4,5,6,7,8,9,10,11,12,13,14,15);
    }
    // B fragments (32x16): lane<16 -> col, K{0..7,16..23} ; lane>=16 -> col, K{8..15,24..31}
    v16bf bf[4];
#pragma unroll
    for (int ni = 0; ni < 4; ++ni) {
      const __bf16* q = &Bs[cur][(waveN * 64 + ni * 16 + hl) * BSTR + hi * 8];
      v8bf lo  = *(const v8bf*)q;
      v8bf hi8 = *(const v8bf*)(q + 16);
      bf[ni] = __builtin_shufflevector(lo, hi8, 0,1,2,3,4,5,6,7,8,9,10,11,12,13,14,15);
    }
#pragma unroll
    for (int mi = 0; mi < 2; ++mi)
#pragma unroll
      for (int ni = 0; ni < 4; ++ni)
        acc[mi][ni] = __builtin_amdgcn_wmma_f32_16x16x32_bf16(
            false, af[mi], false, bf[ni], (short)0, acc[mi][ni], false, false);

    if (kb + 1 < KSTEPS) {
#if USE_ASYNC_COPY
      wait_async0();
      __syncthreads();
#else
      __syncthreads();
      stage((kb + 1) & 1);
      __syncthreads();
#endif
    }
  }

  // epilogue: add per-batch/per-gate bias, store f32 Z
#pragma unroll
  for (int mi = 0; mi < 2; ++mi) {
#pragma unroll
    for (int ni = 0; ni < 4; ++ni) {
      const int col = colBase + waveN * 64 + ni * 16 + hl;
      const int g = col >> 9, j = col & 511;
#pragma unroll
      for (int v = 0; v < 8; ++v) {
        const long row = rowBase + waveM * 32 + mi * 16 + v + hi * 8;
        const int bb = (int)(row >> 9);
        Z[row * NC + col] = acc[mi][ni][v] + biasE[(((long)bb * 9 + g) << 9) + j];
      }
    }
  }
}

// softmax over S+1 of [sigmoid(z8)+mask_score ; gd]  ->  new dummy c/h
__global__ __launch_bounds__(256) void scores_reduce(
    const float* __restrict__ Z, const float* __restrict__ c,
    const float* __restrict__ gd, const float* __restrict__ go,
    const float* __restrict__ dc, const int* __restrict__ mask,
    float* __restrict__ dh_new, float* __restrict__ dc_new)
{
  int idx = blockIdx.x * 256 + threadIdx.x;            // BATCH*HD
  int b = idx >> 9, j = idx & 511;
  float se = 0.0f, sc = 0.0f;
  for (int s = 0; s < SEQ; ++s) {
    long n = (long)b * SEQ + s;
    float mz = (float)mask[n] * 1e25f - 1e25f;
    float gf = sigmoidf_(Z[n * NC + 8 * 512 + j]) + mz;
    float e = __expf(gf);
    se += e;
    sc += e * c[n * HD + j];
  }
  float ed = __expf(gd[idx]);
  float dcn = (sc + ed * dc[idx]) / (se + ed);
  dc_new[idx] = dcn;
  dh_new[idx] = go[idx] * tanhf(dcn);
}

// gate nonlinearities + cell/hidden update
__global__ __launch_bounds__(256) void update_hc(
    const float* __restrict__ Z, const float* __restrict__ c,
    const float* __restrict__ dc, const int* __restrict__ pos,
    const int* __restrict__ mask, float* __restrict__ h_new, float* __restrict__ c_new)
{
  long idx = (long)blockIdx.x * 256 + threadIdx.x;     // NROW*HD
  long n = idx >> 9; int j = (int)(idx & 511);
  int b = (int)(n >> 9), s = (int)(n & 511);
  const float* zr = Z + n * NC;
  float e[6]; float sum = 0.0f;
#pragma unroll
  for (int g = 0; g < 6; ++g) {
    e[g] = __expf(sigmoidf_(zr[g * 512 + j]));
    sum += e[g];
  }
  float o = sigmoidf_(zr[6 * 512 + j]);
  float u = tanhf(zr[7 * 512 + j]);
  float cf = c[idx];
  float cb = 0.0f, ca = 0.0f;
  if (s >= 1)        cb += c[idx - HD];
  if (s >= 2)        cb += c[idx - 2 * HD];
  if (s <= SEQ - 2)  ca += c[idx + HD];
  if (s <= SEQ - 3)  ca += c[idx + 2 * HD];
  int p = pos[n];
  float swc = (p == 0) ? 0.0f : c[((long)b * SEQ + (p - 1)) * HD + j];
  float tdc = dc[b * 512 + j];
  float cn = (e[0] * cb + e[1] * ca + e[2] * cf + e[3] * tdc + e[4] * swc + e[5] * u) / sum;
  float m = (float)mask[n];
  c_new[idx] = cn * m;
  h_new[idx] = (o * tanhf(cn)) * m;
}

// ---------------------------------------------------------------------------
extern "C" void kernel_launch(void* const* d_in, const int* in_sizes, int n_in,
                              void* d_out, int out_size, void* d_ws, size_t ws_size,
                              hipStream_t stream) {
  (void)in_sizes; (void)n_in; (void)out_size; (void)ws_size;
  const float* word  = (const float*)d_in[0];
  const float* ih    = (const float*)d_in[1];
  const float* ic    = (const float*)d_in[2];
  const float* Wx    = (const float*)d_in[3];
  const float* Wh    = (const float*)d_in[4];
  const float* Wi    = (const float*)d_in[5];
  const float* Wd    = (const float*)d_in[6];
  const float* Ws    = (const float*)d_in[7];
  const float* bvec  = (const float*)d_in[8];
  const float* gW    = (const float*)d_in[9];
  const float* gb    = (const float*)d_in[10];
  const int*   pos   = (const int*)d_in[11];
  const int*   mask  = (const int*)d_in[12];
  const int    num_layers = 4;                         // matches reference setup

  uintptr_t base = (uintptr_t)d_ws;
  size_t off = 0;
  auto take = [&](size_t bytes) -> void* {
    off = (off + 255) & ~(size_t)255;
    void* p = (void*)(base + off);
    off += bytes;
    return p;
  };
  __bf16* Bpack = (__bf16*)take((size_t)KA * NC * 2);
  __bf16* Abig  = (__bf16*)take((size_t)NROW * KA * 2);
  float*  Z     = (float*) take((size_t)NROW * NC * 4);
  float*  hb0   = (float*) take((size_t)NROW * HD * 4);
  float*  hb1   = (float*) take((size_t)NROW * HD * 4);
  float*  cb0   = (float*) take((size_t)NROW * HD * 4);
  float*  cb1   = (float*) take((size_t)NROW * HD * 4);
  float*  dh0   = (float*) take((size_t)BATCH * HD * 4);
  float*  dh1   = (float*) take((size_t)BATCH * HD * 4);
  float*  dc0   = (float*) take((size_t)BATCH * HD * 4);
  float*  dc1   = (float*) take((size_t)BATCH * HD * 4);
  float*  comb  = (float*) take((size_t)BATCH * HD * 4);
  float*  gdv   = (float*) take((size_t)BATCH * HD * 4);
  float*  gov   = (float*) take((size_t)BATCH * HD * 4);
  float*  biasE = (float*) take((size_t)BATCH * 9 * 512 * 4);

  float* hbuf[2] = {hb0, hb1};
  float* cbuf[2] = {cb0, cb1};
  float* dhb[2]  = {dh0, dh1};
  float* dcb[2]  = {dc0, dc1};

  const long NH = NROW * HD;
  const int BH  = BATCH * HD;

  pack_weights<<<(unsigned)(((long)KA * NC + 255) / 256), 256, 0, stream>>>(Wx, Wh, Wi, Ws, gW, Bpack);
  init_hc<<<(unsigned)((NH + 255) / 256), 256, 0, stream>>>(ih, ic, mask, hbuf[0], cbuf[0]);
  init_emb<<<(unsigned)((NROW * (long)WD + 255) / 256), 256, 0, stream>>>(word, mask, Abig);
  mean_bh<<<(BH + 255) / 256, 256, 0, stream>>>(hbuf[0], dhb[0]);
  mean_bh<<<(BH + 255) / 256, 256, 0, stream>>>(cbuf[0], dcb[0]);

  int cur = 0;
  for (int li = 0; li < num_layers; ++li) {
    int nxt = cur ^ 1;
    mean_bh<<<(BH + 255) / 256, 256, 0, stream>>>(hbuf[cur], comb);
    prep_A<<<(unsigned)((NH + 255) / 256), 256, 0, stream>>>(hbuf[cur], pos, Abig);
    small_gates<<<BATCH, 512, 0, stream>>>(dhb[cur], comb, gW, gb, bvec, Wd, gdv, gov, biasE);
    gemm_wmma<<<dim3((unsigned)(NROW / 128), NC / 128), 256, 0, stream>>>(Abig, Bpack, biasE, Z);
    scores_reduce<<<(BH + 255) / 256, 256, 0, stream>>>(Z, cbuf[cur], gdv, gov, dcb[cur], mask, dhb[nxt], dcb[nxt]);
    update_hc<<<(unsigned)((NH + 255) / 256), 256, 0, stream>>>(Z, cbuf[cur], dcb[cur], pos, mask, hbuf[nxt], cbuf[nxt]);
    cur = nxt;
  }
  (void)hipMemcpyAsync(d_out, hbuf[cur], (size_t)NH * 4, hipMemcpyDeviceToDevice, stream);
}
